// FusedDecoderLayer_84473416778313
// MI455X (gfx1250) — compile-verified, matrix operand order checked by reading, not run on hardware
//
#include <hip/hip_runtime.h>

// ---- fixed problem shape (from setup_inputs; chunk_start/end are constant) ----
#define S_LEN  4096
#define HDIM   4096
#define NHEAD  32
#define NKVH   8
#define DH     128
#define IDIM   14336
#define CS     2048
#define LCHUNK 2048

typedef __attribute__((ext_vector_type(16))) __bf16 v16bf;
typedef __attribute__((ext_vector_type(8)))  float  v8f;

__device__ __forceinline__ unsigned short f32_bf16(float f) {
  unsigned int u = __float_as_uint(f);
  u += 0x7FFFu + ((u >> 16) & 1u);          // round-to-nearest-even
  return (unsigned short)(u >> 16);
}

__device__ __forceinline__ v8f v8f_zero() {
  v8f z;
  #pragma unroll
  for (int i = 0; i < 8; i++) z[i] = 0.0f;
  return z;
}

union Frag { v16bf v; uint4 q[2]; };

// bf16 WMMA A/B fragment: lane holds row/col (lane&15); K chunks
// [half*8, half*8+8) and [16+half*8, 16+half*8+8)  (half = lane>>4).
__device__ __forceinline__ v16bf load_frag16(const unsigned short* base, int half) {
  Frag f;
  f.q[0] = *reinterpret_cast<const uint4*>(base + half * 8);
  f.q[1] = *reinterpret_cast<const uint4*>(base + 16 + half * 8);
  return f.v;
}

#define WMMA_BF16(A, B, C) \
  __builtin_amdgcn_wmma_f32_16x16x32_bf16(false, (A), false, (B), (short)0, (C), false, false)

// CDNA5 async global->LDS copy (16B per lane), tracked by ASYNCcnt.
// lds_off: LDS byte offset (low 32 bits of generic shared pointer, per aperture layout).
__device__ __forceinline__ void async_b128(unsigned lds_off, const void* gptr) {
  asm volatile("global_load_async_to_lds_b128 %0, %1, off"
               :: "v"(lds_off), "v"(gptr) : "memory");
}
__device__ __forceinline__ void wait_async0() {
  asm volatile("s_wait_asynccnt 0" ::: "memory");
}
__device__ __forceinline__ unsigned lds_off(const void* p) {
  return (unsigned)(unsigned long long)(size_t)p;
}

__device__ __forceinline__ uint4 pack8(float4 a, float4 b) {
  uint4 r;
  r.x = (unsigned)f32_bf16(a.x) | ((unsigned)f32_bf16(a.y) << 16);
  r.y = (unsigned)f32_bf16(a.z) | ((unsigned)f32_bf16(a.w) << 16);
  r.z = (unsigned)f32_bf16(b.x) | ((unsigned)f32_bf16(b.y) << 16);
  r.w = (unsigned)f32_bf16(b.z) | ((unsigned)f32_bf16(b.w) << 16);
  return r;
}

__device__ __forceinline__ float block_reduce_sum256(float v, float* red) {
  #pragma unroll
  for (int off = 16; off > 0; off >>= 1) v += __shfl_xor(v, off, 32);
  int lane = threadIdx.x & 31, wid = threadIdx.x >> 5;
  if (lane == 0) red[wid] = v;
  __syncthreads();
  if (wid == 0) {
    float t = (lane < 8) ? red[lane] : 0.0f;
    #pragma unroll
    for (int off = 4; off > 0; off >>= 1) t += __shfl_xor(t, off, 32);
    if (lane == 0) red[0] = t;
  }
  __syncthreads();
  return red[0];
}

// ---------------------------------------------------------------- RMSNorm -> bf16
__global__ __launch_bounds__(256)
void rmsnorm_bf16_kernel(const float* __restrict__ x, const float* __restrict__ w,
                         unsigned short* __restrict__ out, int cols) {
  __shared__ float red[8];
  int row = blockIdx.x;
  const float* xr = x + (size_t)row * cols;
  float ss = 0.0f;
  for (int c = threadIdx.x; c < cols; c += 256) { float v = xr[c]; ss += v * v; }
  float tot = block_reduce_sum256(ss, red);
  float inv = rsqrtf(tot / (float)cols + 1e-5f);
  for (int c = threadIdx.x; c < cols; c += 256)
    out[(size_t)row * cols + c] = f32_bf16(xr[c] * inv * w[c]);
}

// ---------------------------------------------------------------- generic WMMA GEMM
// C[M,N] = A_bf16[M,K] * B_f32[N,K]^T   (weights are [out,in] row-major)
#define BM 128
#define BN 128
#define BK 64
#define LDT 72   // LDS pitch in ushorts (72*2B = 144B, multiple of 16B)

__global__ __launch_bounds__(256)
void gemm_bf16_kernel(const unsigned short* __restrict__ A,
                      const float* __restrict__ B,
                      float* __restrict__ C,
                      int M, int N, int K) {
  __shared__ unsigned short sA[BM * LDT];
  __shared__ unsigned short sB[BN * LDT];
  const int bn = blockIdx.x, bm = blockIdx.y;
  const int tid = threadIdx.x;
  const int wave = tid >> 5, lane = tid & 31;
  const int wm = wave & 3, wn = wave >> 2;     // 4x2 waves -> each wave 32x64
  const int half = lane >> 4, l16 = lane & 15;
  (void)M;

  v8f acc[2][4];
  #pragma unroll
  for (int i = 0; i < 2; i++)
    #pragma unroll
    for (int j = 0; j < 4; j++) acc[i][j] = v8f_zero();

  const int ar = tid >> 1, ac = (tid & 1) * 32;   // A: 128 rows x 64 k, 32 elems/thr
  const int br = tid >> 1, bc = (tid & 1) * 32;   // B: 128 rows x 64 k, 32 f32/thr
  const unsigned short* ag = A + (size_t)(bm * BM + ar) * K + ac;
  const float* bg = B + (size_t)(bn * BN + br) * K + bc;
  const unsigned aL = lds_off(sA + ar * LDT + ac);
  unsigned short* bL = sB + br * LDT + bc;

  for (int k0 = 0; k0 < K; k0 += BK) {
    // stage A (bf16 global -> LDS, async DMA path)
    #pragma unroll
    for (int i = 0; i < 4; i++)
      async_b128(aL + i * 16, ag + k0 + i * 8);
    // stage B with f32->bf16 conversion through VGPRs
    {
      const float4* g = reinterpret_cast<const float4*>(bg + k0);
      #pragma unroll
      for (int i = 0; i < 4; i++)
        *reinterpret_cast<uint4*>(bL + i * 8) = pack8(g[2 * i], g[2 * i + 1]);
    }
    wait_async0();
    __syncthreads();
    #pragma unroll
    for (int kk = 0; kk < 2; kk++) {
      const unsigned short* pa = sA + (wm * 32 + l16) * LDT + kk * 32;
      v16bf a0 = load_frag16(pa, half);
      v16bf a1 = load_frag16(pa + 16 * LDT, half);
      #pragma unroll
      for (int j = 0; j < 4; j++) {
        v16bf b = load_frag16(sB + (wn * 64 + j * 16 + l16) * LDT + kk * 32, half);
        acc[0][j] = WMMA_BF16(a0, b, acc[0][j]);
        acc[1][j] = WMMA_BF16(a1, b, acc[1][j]);
      }
    }
    __syncthreads();
  }

  #pragma unroll
  for (int i = 0; i < 2; i++)
    #pragma unroll
    for (int j = 0; j < 4; j++) {
      int row0 = bm * BM + wm * 32 + i * 16 + 8 * half;
      int col  = bn * BN + wn * 64 + j * 16 + l16;
      float* cp = C + (size_t)row0 * N + col;
      #pragma unroll
      for (int r = 0; r < 8; r++) cp[(size_t)r * N] = acc[i][j][r];
    }
}

// ---------------------------------------------------------------- fused gate/up -> silu(g)*u bf16
#define GBN 64

__global__ __launch_bounds__(256)
void gateup_kernel(const unsigned short* __restrict__ A,
                   const float* __restrict__ Bg,
                   const float* __restrict__ Bu,
                   unsigned short* __restrict__ act,
                   int M, int N, int K) {
  __shared__ unsigned short sA[BM * LDT];
  __shared__ unsigned short sG[GBN * LDT];
  __shared__ unsigned short sU[GBN * LDT];
  const int bn = blockIdx.x, bm = blockIdx.y;
  const int tid = threadIdx.x;
  const int wave = tid >> 5, lane = tid & 31;
  const int wm = wave & 3, wn = wave >> 2;     // each wave 32x32 (for g and u)
  const int half = lane >> 4, l16 = lane & 15;
  (void)M;

  v8f ag2[2][2], au2[2][2];
  #pragma unroll
  for (int i = 0; i < 2; i++)
    #pragma unroll
    for (int j = 0; j < 2; j++) { ag2[i][j] = v8f_zero(); au2[i][j] = v8f_zero(); }

  const int ar = tid >> 1, ac = (tid & 1) * 32;
  const int br = tid >> 2, bc = (tid & 3) * 16;   // 64 rows x 64 k, 16 f32/thr/matrix
  const unsigned short* agp = A + (size_t)(bm * BM + ar) * K + ac;
  const float* bgg = Bg + (size_t)(bn * GBN + br) * K + bc;
  const float* bgu = Bu + (size_t)(bn * GBN + br) * K + bc;
  const unsigned aL = lds_off(sA + ar * LDT + ac);
  unsigned short* gL = sG + br * LDT + bc;
  unsigned short* uL = sU + br * LDT + bc;

  for (int k0 = 0; k0 < K; k0 += BK) {
    #pragma unroll
    for (int i = 0; i < 4; i++)
      async_b128(aL + i * 16, agp + k0 + i * 8);
    {
      const float4* g = reinterpret_cast<const float4*>(bgg + k0);
      const float4* u = reinterpret_cast<const float4*>(bgu + k0);
      #pragma unroll
      for (int i = 0; i < 2; i++) {
        *reinterpret_cast<uint4*>(gL + i * 8) = pack8(g[2 * i], g[2 * i + 1]);
        *reinterpret_cast<uint4*>(uL + i * 8) = pack8(u[2 * i], u[2 * i + 1]);
      }
    }
    wait_async0();
    __syncthreads();
    #pragma unroll
    for (int kk = 0; kk < 2; kk++) {
      const unsigned short* pa = sA + (wm * 32 + l16) * LDT + kk * 32;
      v16bf a0 = load_frag16(pa, half);
      v16bf a1 = load_frag16(pa + 16 * LDT, half);
      const unsigned short* pg = sG + (wn * 32 + l16) * LDT + kk * 32;
      const unsigned short* pu = sU + (wn * 32 + l16) * LDT + kk * 32;
      v16bf g0 = load_frag16(pg, half);
      v16bf g1 = load_frag16(pg + 16 * LDT, half);
      v16bf u0 = load_frag16(pu, half);
      v16bf u1 = load_frag16(pu + 16 * LDT, half);
      ag2[0][0] = WMMA_BF16(a0, g0, ag2[0][0]);
      ag2[0][1] = WMMA_BF16(a0, g1, ag2[0][1]);
      ag2[1][0] = WMMA_BF16(a1, g0, ag2[1][0]);
      ag2[1][1] = WMMA_BF16(a1, g1, ag2[1][1]);
      au2[0][0] = WMMA_BF16(a0, u0, au2[0][0]);
      au2[0][1] = WMMA_BF16(a0, u1, au2[0][1]);
      au2[1][0] = WMMA_BF16(a1, u0, au2[1][0]);
      au2[1][1] = WMMA_BF16(a1, u1, au2[1][1]);
    }
    __syncthreads();
  }

  #pragma unroll
  for (int i = 0; i < 2; i++)
    #pragma unroll
    for (int j = 0; j < 2; j++) {
      int row0 = bm * BM + wm * 32 + i * 16 + 8 * half;
      int col  = bn * GBN + wn * 32 + j * 16 + l16;
      unsigned short* cp = act + (size_t)row0 * N + col;
      #pragma unroll
      for (int r = 0; r < 8; r++) {
        float g = ag2[i][j][r];
        float u = au2[i][j][r];
        float s = g / (1.0f + __expf(-g));    // silu
        cp[(size_t)r * N] = f32_bf16(s * u);
      }
    }
}

// ---------------------------------------------------------------- RoPE / layout kernels
__global__ void rope_q_kernel(const float* __restrict__ x,
                              const float* __restrict__ cosb, const float* __restrict__ sinb,
                              unsigned short* __restrict__ out) {
  int idx = blockIdx.x * blockDim.x + threadIdx.x;
  if (idx >= LCHUNK * NHEAD * DH) return;
  int d = idx & (DH - 1);
  int h = (idx >> 7) & (NHEAD - 1);
  int l = idx >> 12;
  const float* row = x + (size_t)l * (NHEAD * DH) + (size_t)h * DH;
  float v = row[d];
  float p = (d < DH / 2) ? -row[d + DH / 2] : row[d - DH / 2];
  int pos = CS + l;
  float c = cosb[(size_t)pos * DH + d];
  float s = sinb[(size_t)pos * DH + d];
  out[((size_t)h * LCHUNK + l) * DH + d] = f32_bf16(v * c + p * s);
}

__global__ void rope_k_kernel(const float* __restrict__ x,
                              const float* __restrict__ cosb, const float* __restrict__ sinb,
                              unsigned short* __restrict__ out) {
  int idx = blockIdx.x * blockDim.x + threadIdx.x;
  if (idx >= S_LEN * NKVH * DH) return;
  int d = idx & (DH - 1);
  int kv = (idx >> 7) & (NKVH - 1);
  int s = idx >> 10;
  const float* row = x + (size_t)s * (NKVH * DH) + (size_t)kv * DH;
  float v = row[d];
  float p = (d < DH / 2) ? -row[d + DH / 2] : row[d - DH / 2];
  float c = cosb[(size_t)s * DH + d];
  float sn = sinb[(size_t)s * DH + d];
  out[((size_t)kv * S_LEN + s) * DH + d] = f32_bf16(v * c + p * sn);
}

__global__ void perm_v_kernel(const float* __restrict__ x, unsigned short* __restrict__ vt) {
  int idx = blockIdx.x * blockDim.x + threadIdx.x;
  if (idx >= S_LEN * NKVH * DH) return;
  int d = idx & (DH - 1);
  int kv = (idx >> 7) & (NKVH - 1);
  int s = idx >> 10;
  vt[((size_t)kv * DH + d) * S_LEN + s] = f32_bf16(x[(size_t)s * (NKVH * DH) + (size_t)kv * DH + d]);
}

// ---------------------------------------------------------------- flash attention (WMMA)
#define KP 136   // sK pitch: 64 keys x 128 d
#define VP 72    // sV pitch: 128 d x 64 s
#define PP 72    // sP pitch: per-wave 16 x 64

__global__ __launch_bounds__(128)
void flash_attn_kernel(const unsigned short* __restrict__ qb,   // [NH][L][DH]
                       const unsigned short* __restrict__ kbf,  // [NKV][S][DH]
                       const unsigned short* __restrict__ vt,   // [NKV][DH][S]
                       unsigned short* __restrict__ ob) {       // [L][NH*DH]
  __shared__ unsigned short sK[64 * KP];
  __shared__ unsigned short sV[DH * VP];
  __shared__ unsigned short sP[4 * 16 * PP];
  const int qt = blockIdx.x;
  const int h  = blockIdx.y;
  const int kv = h >> 2;                 // G = NHEAD / NKVH = 4
  const int tid = threadIdx.x;
  const int w = tid >> 5, lane = tid & 31;
  const int half = lane >> 4, l16 = lane & 15;
  const int qrow0 = qt * 64 + w * 16;    // this wave's 16 q rows (chunk-local)

  // preload Q fragments: 16 rows x 128 d -> 4 A-fragments
  v16bf qf[4];
  {
    const unsigned short* qbase = qb + ((size_t)h * LCHUNK + qrow0 + l16) * DH;
    #pragma unroll
    for (int kk = 0; kk < 4; kk++) qf[kk] = load_frag16(qbase + kk * 32, half);
  }

  v8f oacc[8];
  #pragma unroll
  for (int n = 0; n < 8; n++) oacc[n] = v8f_zero();
  float mrun[8], lsum[8];
  #pragma unroll
  for (int r = 0; r < 8; r++) { mrun[r] = -3.0e38f; lsum[r] = 0.0f; }

  const int nkb = CS / 64 + qt + 1;      // causal key-block count
  const float scale = 0.088388347648318447f;  // 1/sqrt(128)

  const int krow = tid >> 1, kseg = (tid & 1) * 64;
  const unsigned kL = lds_off(sK + krow * KP + kseg);
  const unsigned vL = lds_off(sV + tid * VP);

  for (int kbi = 0; kbi < nkb; kbi++) {
    __syncthreads();
    { // stage K block (64 keys x 128 d) via async DMA
      const unsigned short* g = kbf + ((size_t)kv * S_LEN + kbi * 64 + krow) * DH + kseg;
      #pragma unroll
      for (int i = 0; i < 8; i++) async_b128(kL + i * 16, g + i * 8);
    }
    { // stage V block (128 d x 64 s) via async DMA
      const unsigned short* g = vt + ((size_t)kv * DH + tid) * S_LEN + kbi * 64;
      #pragma unroll
      for (int i = 0; i < 8; i++) async_b128(vL + i * 16, g + i * 8);
    }
    wait_async0();
    __syncthreads();

    // scores: 16 q rows x 64 keys
    v8f sacc[4];
    #pragma unroll
    for (int j = 0; j < 4; j++) sacc[j] = v8f_zero();
    #pragma unroll
    for (int kk = 0; kk < 4; kk++) {
      #pragma unroll
      for (int j = 0; j < 4; j++) {
        v16bf bk = load_frag16(sK + (j * 16 + l16) * KP + kk * 32, half);
        sacc[j] = WMMA_BF16(qf[kk], bk, sacc[j]);
      }
    }

    // online softmax (rows live across 16 lanes within a half-wave)
    float corr[8];
    #pragma unroll
    for (int r = 0; r < 8; r++) {
      int qpos = CS + qrow0 + r + 8 * half;
      float sv[4];
      #pragma unroll
      for (int j = 0; j < 4; j++) {
        int key = kbi * 64 + j * 16 + l16;
        float xx = sacc[j][r] * scale;
        sv[j] = (key <= qpos) ? xx : -3.0e38f;
      }
      float mx = fmaxf(fmaxf(sv[0], sv[1]), fmaxf(sv[2], sv[3]));
      #pragma unroll
      for (int off = 1; off < 16; off <<= 1) mx = fmaxf(mx, __shfl_xor(mx, off, 32));
      float mn = fmaxf(mrun[r], mx);
      float c  = __expf(mrun[r] - mn);
      mrun[r] = mn;
      float rs = 0.0f;
      #pragma unroll
      for (int j = 0; j < 4; j++) { float p = __expf(sv[j] - mn); sv[j] = p; rs += p; }
      #pragma unroll
      for (int off = 1; off < 16; off <<= 1) rs += __shfl_xor(rs, off, 32);
      lsum[r] = lsum[r] * c + rs;
      corr[r] = c;
      #pragma unroll
      for (int j = 0; j < 4; j++)
        sP[(w * 16 + r + 8 * half) * PP + j * 16 + l16] = f32_bf16(sv[j]);
    }

    #pragma unroll
    for (int n = 0; n < 8; n++)
      #pragma unroll
      for (int r = 0; r < 8; r++) oacc[n][r] *= corr[r];

    // same-wave LDS RAW on sP: ensure stores land before fragment reloads
    asm volatile("s_wait_dscnt 0" ::: "memory");

    // P (16x64) x V (64x128)
    const unsigned short* sPw = sP + w * 16 * PP;
    #pragma unroll
    for (int kk = 0; kk < 2; kk++) {
      v16bf pa = load_frag16(sPw + l16 * PP + kk * 32, half);
      #pragma unroll
      for (int n = 0; n < 8; n++) {
        v16bf vb = load_frag16(sV + (n * 16 + l16) * VP + kk * 32, half);
        oacc[n] = WMMA_BF16(pa, vb, oacc[n]);
      }
    }
  }

  // normalize and write o (bf16, [L][NH*DH])
  #pragma unroll
  for (int n = 0; n < 8; n++) {
    #pragma unroll
    for (int r = 0; r < 8; r++) {
      int row = qrow0 + r + 8 * half;
      ob[(size_t)row * (NHEAD * DH) + (size_t)h * DH + n * 16 + l16] =
          f32_bf16(oacc[n][r] / lsum[r]);
    }
  }
}

// ---------------------------------------------------------------- residual + RMSNorm2
__global__ __launch_bounds__(256)
void resid_rms_kernel(const float* __restrict__ hs, const float* __restrict__ attn,
                      const float* __restrict__ w, float* __restrict__ hidden,
                      unsigned short* __restrict__ h2) {
  __shared__ float red[8];
  int row = blockIdx.x;
  const float* a = hs + (size_t)(CS + row) * HDIM;
  const float* b = attn + (size_t)row * HDIM;
  float v[16];
  float ss = 0.0f;
  #pragma unroll
  for (int i = 0; i < 16; i++) {
    int c = threadIdx.x + i * 256;
    float x = a[c] + b[c];
    v[i] = x; ss += x * x;
  }
  float tot = block_reduce_sum256(ss, red);
  float inv = rsqrtf(tot / (float)HDIM + 1e-5f);
  #pragma unroll
  for (int i = 0; i < 16; i++) {
    int c = threadIdx.x + i * 256;
    hidden[(size_t)row * HDIM + c] = v[i];
    h2[(size_t)row * HDIM + c] = f32_bf16(v[i] * inv * w[c]);
  }
}

__global__ void final_add_kernel(const float* __restrict__ hidden,
                                 const float* __restrict__ mlp,
                                 float* __restrict__ out) {
  int idx = blockIdx.x * blockDim.x + threadIdx.x;
  if (idx >= LCHUNK * HDIM) return;
  out[idx] = hidden[idx] + mlp[idx];
}

// ---------------------------------------------------------------- launch
extern "C" void kernel_launch(void* const* d_in, const int* in_sizes, int n_in,
                              void* d_out, int out_size, void* d_ws, size_t ws_size,
                              hipStream_t stream) {
  (void)in_sizes; (void)n_in; (void)out_size; (void)ws_size;

  const float* hs     = (const float*)d_in[0];
  const float* cosb   = (const float*)d_in[1];
  const float* sinb   = (const float*)d_in[2];
  const float* q_w    = (const float*)d_in[3];
  const float* k_w    = (const float*)d_in[4];
  const float* v_w    = (const float*)d_in[5];
  const float* o_w    = (const float*)d_in[6];
  const float* gate_w = (const float*)d_in[7];
  const float* up_w   = (const float*)d_in[8];
  const float* down_w = (const float*)d_in[9];
  const float* ln1    = (const float*)d_in[10];
  const float* ln2    = (const float*)d_in[11];

  char* wp = (char*)d_ws;
  auto alloc = [&](size_t bytes) -> void* {
    void* p = (void*)wp;
    wp += (bytes + 255) & ~(size_t)255;
    return p;
  };
  unsigned short* h_bf   = (unsigned short*)alloc((size_t)S_LEN * HDIM * 2);
  float*          q_raw  = (float*)alloc((size_t)LCHUNK * NHEAD * DH * 4);
  float*          k_raw  = (float*)alloc((size_t)S_LEN * NKVH * DH * 4);
  float*          v_raw  = (float*)alloc((size_t)S_LEN * NKVH * DH * 4);
  unsigned short* q_bf   = (unsigned short*)alloc((size_t)LCHUNK * NHEAD * DH * 2);
  unsigned short* k_bf   = (unsigned short*)alloc((size_t)S_LEN * NKVH * DH * 2);
  unsigned short* vt_bf  = (unsigned short*)alloc((size_t)S_LEN * NKVH * DH * 2);
  unsigned short* o_bf   = (unsigned short*)alloc((size_t)LCHUNK * HDIM * 2);
  float*          attn_o = (float*)alloc((size_t)LCHUNK * HDIM * 4);
  float*          hidden = (float*)alloc((size_t)LCHUNK * HDIM * 4);
  unsigned short* h2_bf  = (unsigned short*)alloc((size_t)LCHUNK * HDIM * 2);
  unsigned short* act_bf = (unsigned short*)alloc((size_t)LCHUNK * IDIM * 2);
  float*          mlp    = (float*)alloc((size_t)LCHUNK * HDIM * 4);

  // 1) h = rmsnorm(hidden_states, ln1) -> bf16  (all S rows)
  rmsnorm_bf16_kernel<<<S_LEN, 256, 0, stream>>>(hs, ln1, h_bf, HDIM);

  // 2) Q/K/V projections
  gemm_bf16_kernel<<<dim3((NHEAD * DH) / BN, LCHUNK / BM), 256, 0, stream>>>(
      h_bf + (size_t)CS * HDIM, q_w, q_raw, LCHUNK, NHEAD * DH, HDIM);
  gemm_bf16_kernel<<<dim3((NKVH * DH) / BN, S_LEN / BM), 256, 0, stream>>>(
      h_bf, k_w, k_raw, S_LEN, NKVH * DH, HDIM);
  gemm_bf16_kernel<<<dim3((NKVH * DH) / BN, S_LEN / BM), 256, 0, stream>>>(
      h_bf, v_w, v_raw, S_LEN, NKVH * DH, HDIM);

  // 3) RoPE + layout
  rope_q_kernel<<<(LCHUNK * NHEAD * DH + 255) / 256, 256, 0, stream>>>(q_raw, cosb, sinb, q_bf);
  rope_k_kernel<<<(S_LEN * NKVH * DH + 255) / 256, 256, 0, stream>>>(k_raw, cosb, sinb, k_bf);
  perm_v_kernel<<<(S_LEN * NKVH * DH + 255) / 256, 256, 0, stream>>>(v_raw, vt_bf);

  // 4) flash attention
  flash_attn_kernel<<<dim3(LCHUNK / 64, NHEAD), 128, 0, stream>>>(q_bf, k_bf, vt_bf, o_bf);

  // 5) output projection
  gemm_bf16_kernel<<<dim3(HDIM / BN, LCHUNK / BM), 256, 0, stream>>>(
      o_bf, o_w, attn_o, LCHUNK, HDIM, HDIM);

  // 6) residual + rmsnorm2
  resid_rms_kernel<<<LCHUNK, 256, 0, stream>>>(hs, attn_o, ln2, hidden, h2_bf);

  // 7) fused gate/up -> silu(g)*u bf16
  gateup_kernel<<<dim3(IDIM / GBN, LCHUNK / BM), 256, 0, stream>>>(
      h2_bf, gate_w, up_w, act_bf, LCHUNK, IDIM, HDIM);

  // 8) down projection
  gemm_bf16_kernel<<<dim3(HDIM / BN, LCHUNK / BM), 256, 0, stream>>>(
      act_bf, down_w, mlp, LCHUNK, HDIM, IDIM);

  // 9) final residual add
  final_add_kernel<<<(LCHUNK * HDIM + 255) / 256, 256, 0, stream>>>(
      hidden, mlp, (float*)d_out);
}
